// InternConvertedInternLMAttention_59596966200071
// MI455X (gfx1250) — compile-verified, matrix operand order checked by reading, not run on hardware
//
#include <hip/hip_runtime.h>
#include <hip/hip_bf16.h>

#define S_LEN  2048
#define D_DIM  4096
#define H_NUM  32
#define HD_DIM 128
#define GM 2048
#define GN 4096
#define GK 4096

typedef __attribute__((ext_vector_type(8)))  __bf16 v8bf;
typedef __attribute__((ext_vector_type(16))) __bf16 v16bf;
typedef __attribute__((ext_vector_type(8)))  float  v8f;
typedef __attribute__((ext_vector_type(4)))  unsigned int v4u;
typedef __attribute__((ext_vector_type(8)))  int v8i;
typedef __attribute__((ext_vector_type(4)))  int v4i;

#if __has_builtin(__builtin_amdgcn_tensor_load_to_lds) && \
    __has_builtin(__builtin_amdgcn_s_wait_tensorcnt)
#define USE_TDM 1
#else
#define USE_TDM 0
#endif

union FragU { struct { v8bf lo, hi; } p; v16bf v; };

// 16-bit WMMA A/B fragment load: lane half holds K = ho..ho+7, 16+ho..16+ho+7
__device__ __forceinline__ v16bf load_frag(const __bf16* __restrict__ base, int ho) {
  FragU f;
  f.p.lo = *(const v8bf*)(base + ho);
  f.p.hi = *(const v8bf*)(base + 16 + ho);
  return f.v;
}

__device__ __forceinline__ v8f wmma_bf16(v16bf a, v16bf b, v8f c) {
  return __builtin_amdgcn_wmma_f32_16x16x32_bf16(false, a, false, b, (short)0, c,
                                                 false, false);
}

__global__ void cvt_f32_to_bf16(const float* __restrict__ in,
                                __bf16* __restrict__ out, int n) {
  int i = blockIdx.x * blockDim.x + threadIdx.x;
  int stride = gridDim.x * blockDim.x;
  for (; i < n; i += stride) out[i] = (__bf16)in[i];
}

// out = X @ W^T + b (bf16 in, f32 out). 2048x4096x4096 fixed.
// Wave tile 64x64 (16 accums, 16 WMMAs / 32-k step), register double-buffered.
// Block = 4 waves (2x2) -> 128x128 block tile.
// launch_bounds(...,1): allow the full wave32 register file (no spills);
// single-wave-per-SIMD is the intended regime for back-to-back WMMA on CDNA5.
__global__ void __launch_bounds__(128, 1)
gemm_xwt_bias(const __bf16* __restrict__ A,
              const __bf16* __restrict__ W,
              const float* __restrict__ bias,
              float* __restrict__ out) {
  const int lane = threadIdx.x & 31;
  const int wave = threadIdx.x >> 5;
  const int r    = lane & 15;
  const int ho   = (lane >> 4) * 8;
  const int m0   = blockIdx.y * 128 + (wave >> 1) * 64;
  const int n0   = blockIdx.x * 128 + (wave & 1) * 64;

  v8f acc[4][4];
#pragma unroll
  for (int i = 0; i < 4; ++i)
#pragma unroll
    for (int j = 0; j < 4; ++j) acc[i][j] = (v8f){};

  const __bf16* arow[4];
  const __bf16* brow[4];
#pragma unroll
  for (int i = 0; i < 4; ++i) {
    arow[i] = A + (size_t)(m0 + i * 16 + r) * GK;
    brow[i] = W + (size_t)(n0 + i * 16 + r) * GK;
  }

  v16bf af[4], bf[4];
#pragma unroll
  for (int i = 0; i < 4; ++i) {
    af[i] = load_frag(arow[i], ho);
    bf[i] = load_frag(brow[i], ho);
  }

  for (int k0 = 0; k0 < GK; k0 += 32) {
    const int kn = (k0 + 32) & (GK - 1);  // wraps to 0 on last iter (dummy reload)
    v16bf an[4], bn[4];
#pragma unroll
    for (int i = 0; i < 4; ++i) {
      an[i] = load_frag(arow[i] + kn, ho);
      bn[i] = load_frag(brow[i] + kn, ho);
    }
#pragma unroll
    for (int i = 0; i < 4; ++i)
#pragma unroll
      for (int j = 0; j < 4; ++j)
        acc[i][j] = wmma_bf16(af[i], bf[j], acc[i][j]);
#pragma unroll
    for (int i = 0; i < 4; ++i) { af[i] = an[i]; bf[i] = bn[i]; }
  }

#pragma unroll
  for (int i = 0; i < 4; ++i)
#pragma unroll
    for (int t = 0; t < 4; ++t) {
      int col = n0 + t * 16 + r;
      float bv = bias[col];
#pragma unroll
      for (int j = 0; j < 8; ++j) {
        int row = m0 + i * 16 + j + ho;
        out[(size_t)row * GN + col] = acc[i][t][j] + bv;
      }
    }
}

// RoPE on fp32 (S,H,HD), emit bf16.
__global__ void rope_cvt(const float* __restrict__ in, __bf16* __restrict__ out) {
  int idx = blockIdx.x * blockDim.x + threadIdx.x;  // S*H*64
  if (idx >= S_LEN * H_NUM * 64) return;
  int i = idx & 63;
  int h = (idx >> 6) & (H_NUM - 1);
  int s = idx >> 11;
  float invf = __powf(10000.0f, -(float)(2 * i) * (1.0f / 128.0f));
  float ang = (float)s * invf;
  float c = __cosf(ang), sn = __sinf(ang);
  size_t base = (size_t)s * D_DIM + h * HD_DIM + i;
  float x1 = in[base], x2 = in[base + 64];
  out[base]      = (__bf16)(x1 * c - x2 * sn);
  out[base + 64] = (__bf16)(x1 * sn + x2 * c);
}

// Fused causal flash attention. One wave per (32-query tile, head): 2 q-tiles
// share every K fragment and V tile. S^T layout keeps softmax lane-local
// (+ one shfl_xor(16)); P^T packs lane-locally into the PV A-fragment.
// V block (32x128 bf16) staged to LDS by the Tensor Data Mover, overlapped
// with the QK WMMAs + softmax; s_wait_tensorcnt before the PV LDS reads.
__global__ void __launch_bounds__(32, 1)
flash_attn(const __bf16* __restrict__ Q,
           const __bf16* __restrict__ Kt,
           const __bf16* __restrict__ V,
           __bf16* __restrict__ ctx) {
  __shared__ __align__(16) __bf16 vlds[32 * 128];
  const int lane = threadIdx.x;
  const int r    = lane & 15;
  const int ho   = (lane >> 4) * 8;
  const int q0   = blockIdx.x * 32;
  const int h    = blockIdx.y;
  const float scale = 0.08838834764831845f;  // 1/sqrt(128)

  v16bf qf[2][4];
#pragma unroll
  for (int qt = 0; qt < 2; ++qt)
#pragma unroll
    for (int c = 0; c < 4; ++c)
      qf[qt][c] =
          load_frag(Q + (size_t)(q0 + qt * 16 + r) * D_DIM + h * HD_DIM + c * 32, ho);

  v8f acc[2][8];
#pragma unroll
  for (int qt = 0; qt < 2; ++qt)
#pragma unroll
    for (int dt = 0; dt < 8; ++dt) acc[qt][dt] = (v8f){};
  float m_prev[2] = {-3.0e38f, -3.0e38f};
  float l[2] = {0.0f, 0.0f};

  const int kend = q0 + 32;  // diagonal block serves both q-tiles via masking

  for (int kb = 0; kb < kend; kb += 32) {
    __syncthreads();  // previous PV LDS reads done before DMA overwrites
#if USE_TDM
    {
      unsigned lds_base = (unsigned)(size_t)(&vlds[0]);
      unsigned long long ga =
          (unsigned long long)(size_t)(V + (size_t)kb * D_DIM + h * HD_DIM);
      v4u g0;
      g0[0] = 1u;                                   // count=1, is_restore=0
      g0[1] = lds_base;                             // lds_addr
      g0[2] = (unsigned)(ga & 0xffffffffu);         // global_addr[31:0]
      g0[3] = (unsigned)((ga >> 32) & 0x1ffffffu)   // global_addr[56:32]
              | (2u << 30);                         // type=2 (image)
      v8i g1;
      g1[0] = 0x00010000;            // wg_mask=0, data_size=1 (2B)
      g1[1] = (int)(128u << 16);     // tensor_dim0[15:0] = 128 (bits 63:48)
      g1[2] = (int)(32u << 16);      // tensor_dim0 hi=0 | tensor_dim1 lo=32
      g1[3] = (int)(128u << 16);     // tensor_dim1 hi=0 | tile_dim0=128
      g1[4] = 32;                    // tile_dim1=32
      g1[5] = 4096;                  // tensor_dim0_stride = D (elements)
      g1[6] = 0;
      g1[7] = 0;
      v4i z4 = {};
      v8i z8 = {};
      // 6-arg form (clang-23 / therock-10.0): (g0, g1, g2, g3, g4, cpol)
      __builtin_amdgcn_tensor_load_to_lds(g0, g1, z4, z4, z8, 0);
    }
#else
#pragma unroll
    for (int i = 0; i < 16; ++i) {
      int chunk = i * 32 + lane;
      int row = chunk >> 4;
      int cc = chunk & 15;
      *(float4*)(&vlds[row * 128 + cc * 8]) =
          *(const float4*)(V + (size_t)(kb + row) * D_DIM + h * HD_DIM + cc * 8);
    }
    __syncthreads();
#endif

    // ---- QK: S^T tiles (32 kk x 32 q), overlaps the TDM transfer ----
    v16bf kf0[4], kf1[4];
#pragma unroll
    for (int c = 0; c < 4; ++c) {
      kf0[c] = load_frag(Kt + (size_t)(kb + r) * D_DIM + h * HD_DIM + c * 32, ho);
      kf1[c] = load_frag(Kt + (size_t)(kb + 16 + r) * D_DIM + h * HD_DIM + c * 32, ho);
    }
    v8f s0[2], s1[2];
#pragma unroll
    for (int qt = 0; qt < 2; ++qt) { s0[qt] = (v8f){}; s1[qt] = (v8f){}; }
#pragma unroll
    for (int c = 0; c < 4; ++c)
#pragma unroll
      for (int qt = 0; qt < 2; ++qt) {
        s0[qt] = wmma_bf16(kf0[c], qf[qt][c], s0[qt]);
        s1[qt] = wmma_bf16(kf1[c], qf[qt][c], s1[qt]);
      }

    // ---- mask + online softmax + pack P^T, per q-tile ----
    v16bf pf[2];
    float alpha[2];
#pragma unroll
    for (int qt = 0; qt < 2; ++qt) {
      float p0[8], p1[8];
      const int qg = q0 + qt * 16 + r;
      float mb = -3.0e38f;
#pragma unroll
      for (int j = 0; j < 8; ++j) {
        int kk0 = kb + ho + j;
        int kk1 = kb + 16 + ho + j;
        float v0 = (kk0 <= qg) ? s0[qt][j] * scale : -3.0e38f;
        float v1 = (kk1 <= qg) ? s1[qt][j] * scale : -3.0e38f;
        p0[j] = v0; p1[j] = v1;
        mb = fmaxf(mb, fmaxf(v0, v1));
      }
      mb = fmaxf(mb, __shfl_xor(mb, 16));
      float m_new = fmaxf(m_prev[qt], mb);
      alpha[qt] = __expf(m_prev[qt] - m_new);
      float lsum = 0.0f;
#pragma unroll
      for (int j = 0; j < 8; ++j) {
        p0[j] = __expf(p0[j] - m_new);
        p1[j] = __expf(p1[j] - m_new);
        lsum += p0[j] + p1[j];
      }
      lsum += __shfl_xor(lsum, 16);
      l[qt] = l[qt] * alpha[qt] + lsum;
      m_prev[qt] = m_new;
      v16bf p = {};
#pragma unroll
      for (int e = 0; e < 8; ++e) {
        p[e]     = (__bf16)p0[e];
        p[e + 8] = (__bf16)p1[e];
      }
      pf[qt] = p;
#pragma unroll
      for (int dt = 0; dt < 8; ++dt)
#pragma unroll
        for (int j = 0; j < 8; ++j) acc[qt][dt][j] *= alpha[qt];
    }

#if USE_TDM
    __builtin_amdgcn_s_wait_tensorcnt(0);
    __syncthreads();
#endif

    // ---- PV: ctx += P(16x32) * V(32x16), V fragments shared by both q-tiles
#pragma unroll
    for (int dt = 0; dt < 8; ++dt) {
      v16bf bfr = {};
#pragma unroll
      for (int e = 0; e < 8; ++e) {
        bfr[e]     = vlds[(ho + e) * 128 + dt * 16 + r];
        bfr[e + 8] = vlds[(16 + ho + e) * 128 + dt * 16 + r];
      }
#pragma unroll
      for (int qt = 0; qt < 2; ++qt)
        acc[qt][dt] = wmma_bf16(pf[qt], bfr, acc[qt][dt]);
    }
  }

#pragma unroll
  for (int qt = 0; qt < 2; ++qt) {
    float inv_l = 1.0f / l[qt];
#pragma unroll
    for (int dt = 0; dt < 8; ++dt)
#pragma unroll
      for (int j = 0; j < 8; ++j) {
        int q = q0 + qt * 16 + j + ho;
        int d = dt * 16 + r;
        ctx[(size_t)q * D_DIM + h * HD_DIM + d] = (__bf16)(acc[qt][dt][j] * inv_l);
      }
  }
}

extern "C" void kernel_launch(void* const* d_in, const int* in_sizes, int n_in,
                              void* d_out, int out_size, void* d_ws, size_t ws_size,
                              hipStream_t stream) {
  (void)in_sizes; (void)n_in; (void)out_size; (void)ws_size;
  const float* X  = (const float*)d_in[0];
  const float* Wq = (const float*)d_in[1];
  const float* bq = (const float*)d_in[2];
  const float* Wk = (const float*)d_in[3];
  const float* bk = (const float*)d_in[4];
  const float* Wv = (const float*)d_in[5];
  const float* bv = (const float*)d_in[6];
  const float* Wo = (const float*)d_in[7];
  const float* bo = (const float*)d_in[8];

  const int S = S_LEN, D = D_DIM;
  char* ws = (char*)d_ws;
  __bf16* Xbf = (__bf16*)ws; ws += (size_t)S * D * 2;
  __bf16* Wbf = (__bf16*)ws; ws += (size_t)D * D * 2;  // reused per weight
  float*  Tf  = (float*)ws;  ws += (size_t)S * D * 4;  // reused per projection
  __bf16* Qbf = (__bf16*)ws; ws += (size_t)S * D * 2;
  __bf16* Kbf = (__bf16*)ws; ws += (size_t)S * D * 2;
  __bf16* Vbf = (__bf16*)ws; ws += (size_t)S * D * 2;
  __bf16* Cbf = (__bf16*)ws; ws += (size_t)S * D * 2;

  const int nSD = S * D, nDD = D * D;
  dim3 gGemm(GN / 128, GM / 128), bGemm(128);
  const int ropeN = S * H_NUM * 64;
  dim3 gRope((ropeN + 255) / 256), bRope(256);
  dim3 gFlash(S / 32, H_NUM), bFlash(32);

  cvt_f32_to_bf16<<<2048, 256, 0, stream>>>(X, Xbf, nSD);

  // Q
  cvt_f32_to_bf16<<<2048, 256, 0, stream>>>(Wq, Wbf, nDD);
  gemm_xwt_bias<<<gGemm, bGemm, 0, stream>>>(Xbf, Wbf, bq, Tf);
  rope_cvt<<<gRope, bRope, 0, stream>>>(Tf, Qbf);

  // K
  cvt_f32_to_bf16<<<2048, 256, 0, stream>>>(Wk, Wbf, nDD);
  gemm_xwt_bias<<<gGemm, bGemm, 0, stream>>>(Xbf, Wbf, bk, Tf);
  rope_cvt<<<gRope, bRope, 0, stream>>>(Tf, Kbf);

  // V
  cvt_f32_to_bf16<<<2048, 256, 0, stream>>>(Wv, Wbf, nDD);
  gemm_xwt_bias<<<gGemm, bGemm, 0, stream>>>(Xbf, Wbf, bv, Tf);
  cvt_f32_to_bf16<<<2048, 256, 0, stream>>>(Tf, Vbf, nSD);

  // attention
  flash_attn<<<gFlash, bFlash, 0, stream>>>(Qbf, Kbf, Vbf, Cbf);

  // output projection -> fp32 d_out
  cvt_f32_to_bf16<<<2048, 256, 0, stream>>>(Wo, Wbf, nDD);
  gemm_xwt_bias<<<gGemm, bGemm, 0, stream>>>(Cbf, Wbf, bo, (float*)d_out);
}
